// TemporalMultiHeadLabelAttentionClassifier_4647154614293
// MI455X (gfx1250) — matrix operand order repaired and checked by reading, not compiled, not run on hardware
//
#include <hip/hip_runtime.h>

// MI455X / gfx1250, wave32. bf16 WMMA (16x16x32, f32 acc) with bf16x2 hi/lo
// split for ~f32 accuracy. Splits are precomputed/materialized so the GEMM
// inner loops are pure b128-load -> v_wmma (no per-use cvt VALU, no exec-mask
// branches).

typedef __attribute__((ext_vector_type(16))) __bf16 v16bf;
typedef __attribute__((ext_vector_type(8)))  __bf16 v8bf;
typedef __attribute__((ext_vector_type(8)))  float  v8f;

namespace {
constexpr int H   = 768;
constexpr int S   = 8192;
constexpr int NH  = 12;
constexpr int DH  = 64;
constexpr int NL  = 50;
constexpr int NLP = 64;   // NL padded to 4 WMMA tiles
constexpr int NC  = 16;
constexpr int T   = 512;

// workspace layout (byte offsets, all 256B aligned)
constexpr size_t SZ_ENC = (size_t)S * H * 2;        // bf16
constexpr size_t SZ_WV  = (size_t)H * H * 2;        // bf16
constexpr size_t SZ_QS  = (size_t)NLP * H * 4;      // f32
constexpr size_t SZ_QK  = (size_t)NH * NLP * H * 2; // bf16
constexpr size_t SZ_VT  = (size_t)H * S * 2;        // bf16
constexpr size_t SZ_LG  = (size_t)NH * NLP * S * 4; // f32
constexpr size_t SZ_MD  = (size_t)NH * NLP * NC * 4;
constexpr size_t SZ_CTX = (size_t)NC * NLP * H * 4;
constexpr size_t SZ_WT  = (size_t)NLP * H * 4;

constexpr size_t B_ENCH = 0;
constexpr size_t B_ENCL = B_ENCH + SZ_ENC;
constexpr size_t B_WVH  = B_ENCL + SZ_ENC;
constexpr size_t B_WVL  = B_WVH + SZ_WV;
constexpr size_t B_QS   = B_WVL + SZ_WV;
constexpr size_t B_QKH  = B_QS + SZ_QS;
constexpr size_t B_QKL  = B_QKH + SZ_QK;
constexpr size_t B_VTH  = B_QKL + SZ_QK;
constexpr size_t B_VTL  = B_VTH + SZ_VT;
constexpr size_t B_LG   = B_VTL + SZ_VT;
constexpr size_t B_MP   = B_LG + SZ_LG;
constexpr size_t B_DP   = B_MP + SZ_MD;
constexpr size_t B_CTX  = B_DP + SZ_MD;
constexpr size_t B_WT   = B_CTX + SZ_CTX;
} // namespace

__device__ __forceinline__ v8f wmma_bf16x3(v16bf ah, v16bf al, v16bf bh, v16bf bl, v8f c) {
  c = __builtin_amdgcn_wmma_f32_16x16x32_bf16(false, ah, false, bh, (short)0, c, false, false);
  c = __builtin_amdgcn_wmma_f32_16x16x32_bf16(false, ah, false, bl, (short)0, c, false, false);
  c = __builtin_amdgcn_wmma_f32_16x16x32_bf16(false, al, false, bh, (short)0, c, false, false);
  return c;
}

// 8 consecutive f32 -> bf16 hi/lo at v16bf elements [base..base+7], masked
__device__ __forceinline__ void cvt8(const float* p, float msk, v16bf& hi, v16bf& lo, int base) {
  float4 a = *(const float4*)p;
  float4 b = *(const float4*)(p + 4);
  float v[8] = {a.x, a.y, a.z, a.w, b.x, b.y, b.z, b.w};
#pragma unroll
  for (int j = 0; j < 8; ++j) {
    float x = v[j] * msk;
    __bf16 h = (__bf16)x;
    hi[base + j] = h;
    lo[base + j] = (__bf16)(x - (float)h);
  }
}

// ---- f32-source fragment loaders (small kernels only) -----------------------
// A frag 16x32 (row-major f32). rows >= mlim contribute 0; no exec branches.
__device__ __forceinline__ void load_a_f32(const float* A, int lda, int m0, int mlim,
                                           int k0, v16bf& hi, v16bf& lo) {
  int lane = threadIdx.x & 31, g = lane >> 4;
  int m = m0 + (lane & 15);
  bool ok = (m < mlim);
  float msk = ok ? 1.f : 0.f;
  const float* row = A + (size_t)(ok ? m : m0) * lda + k0 + (g << 4);
  cvt8(row, msk, hi, lo, 0);
  cvt8(row + 8, msk, hi, lo, 8);
}

// B frag 32x16, source N-major f32: elem(k,n) = Bt[n*ldb + k]
__device__ __forceinline__ void load_b_f32_nmaj(const float* Bt, int ldb, int n0, int k0,
                                                v16bf& hi, v16bf& lo) {
  int lane = threadIdx.x & 31, g = lane >> 4;
  const float* row = Bt + (size_t)(n0 + (lane & 15)) * ldb + k0 + (g << 4);
  cvt8(row, 1.f, hi, lo, 0);
  cvt8(row + 8, 1.f, hi, lo, 8);
}

// B frag 32x16, source K-major f32: elem(k,n) = B[k*ldk + n]
__device__ __forceinline__ void load_b_f32_kmaj(const float* B, int ldk, int n0, int k0,
                                                v16bf& hi, v16bf& lo) {
  int lane = threadIdx.x & 31, g = lane >> 4;
  int n = n0 + (lane & 15);
#pragma unroll
  for (int j = 0; j < 16; ++j) {
    int k = k0 + (g << 4) + j;
    float x = B[(size_t)k * ldk + n];
    __bf16 h = (__bf16)x;
    hi[j] = h;
    lo[j] = (__bf16)(x - (float)h);
  }
}

// ---- bf16-source fragment loaders (hot kernels) -----------------------------
__device__ __forceinline__ v16bf load_a_bf(const __bf16* A, int lda, int m0, int k0) {
  int lane = threadIdx.x & 31, g = lane >> 4;
  const __bf16* row = A + (size_t)(m0 + (lane & 15)) * lda + k0 + (g << 4);
  v8bf p = *(const v8bf*)row;         // K = g*16 + 0..7
  v8bf q = *(const v8bf*)(row + 8);   // K = g*16 + 8 + 0..7
  v16bf r;
#pragma unroll
  for (int j = 0; j < 8; ++j) { r[j] = p[j]; r[8 + j] = q[j]; }
  return r;
}

__device__ __forceinline__ v16bf load_b_bf(const __bf16* Bt, int ldb, int n0, int k0) {
  int lane = threadIdx.x & 31, g = lane >> 4;
  const __bf16* row = Bt + (size_t)(n0 + (lane & 15)) * ldb + k0 + (g << 4);
  return *(const v16bf*)row;          // K = g*16 + 0..15
}

// ---------------- Kernel 0: f32 -> (bf16 hi, bf16 lo), 4 elems/thread
__global__ void k_split4(const float* __restrict__ src, __bf16* __restrict__ hi,
                         __bf16* __restrict__ lo, int n4) {
  int i = blockIdx.x * blockDim.x + threadIdx.x;
  if (i >= n4) return;
  float4 v = ((const float4*)src)[i];
  float vv[4] = {v.x, v.y, v.z, v.w};
#pragma unroll
  for (int j = 0; j < 4; ++j) {
    float x = vv[j];
    __bf16 h = (__bf16)x;
    hi[i * 4 + j] = h;
    lo[i * 4 + j] = (__bf16)(x - (float)h);
  }
}

// ---------------- Kernel 1: Qs = (LQ @ wq^T + bq) * scale, padded to 64 rows
__global__ void k_qproj(const float* __restrict__ lq, const float* __restrict__ inw,
                        const float* __restrict__ inb, float* __restrict__ Qs) {
  int wid = (blockIdx.x * blockDim.x + threadIdx.x) >> 5;  // 192 tiles
  int ntile = wid % (H / 16);
  int mtile = wid / (H / 16);
  int m0 = mtile * 16, n0 = ntile * 16;
  v8f acc = {0.f, 0.f, 0.f, 0.f, 0.f, 0.f, 0.f, 0.f};
  for (int k0 = 0; k0 < H; k0 += 32) {
    v16bf ah, al, bh, bl;
    load_a_f32(lq, H, m0, NL, k0, ah, al);
    load_b_f32_nmaj(inw, H, n0, k0, bh, bl);   // wq rows 0..767
    acc = wmma_bf16x3(ah, al, bh, bl, acc);
  }
  int lane = threadIdx.x & 31, g = lane >> 4, nn = lane & 15;
  float bias = inb[n0 + nn];
#pragma unroll
  for (int r = 0; r < 8; ++r) {
    int m = m0 + r + (g << 3);
    Qs[(size_t)m * H + n0 + nn] = (m < NL) ? (acc[r] + bias) * 0.125f : 0.f;
  }
}

// ---------------- Kernel 2: Vt[d'][s] = enc @ wv^T + bv (bf16 hi/lo, transposed)
__global__ void k_vproj(const __bf16* __restrict__ ench, const __bf16* __restrict__ encl,
                        const __bf16* __restrict__ wvh, const __bf16* __restrict__ wvl,
                        const float* __restrict__ inb,
                        __bf16* __restrict__ Vth, __bf16* __restrict__ Vtl) {
  int wid = (blockIdx.x * blockDim.x + threadIdx.x) >> 5;  // 512*48 tiles
  int ntile = wid % (H / 16);
  int mtile = wid / (H / 16);
  int m0 = mtile * 16, n0 = ntile * 16;
  v8f acc = {0.f, 0.f, 0.f, 0.f, 0.f, 0.f, 0.f, 0.f};
  for (int k0 = 0; k0 < H; k0 += 32) {
    v16bf ah = load_a_bf(ench, H, m0, k0);
    v16bf al = load_a_bf(encl, H, m0, k0);
    v16bf bh = load_b_bf(wvh, H, n0, k0);
    v16bf bl = load_b_bf(wvl, H, n0, k0);
    acc = wmma_bf16x3(ah, al, bh, bl, acc);
  }
  int lane = threadIdx.x & 31, g = lane >> 4, nn = lane & 15;
  int n = n0 + nn;
  float bias = inb[2 * H + n];
#pragma unroll
  for (int r = 0; r < 8; ++r) {
    int m = m0 + r + (g << 3);
    float x = acc[r] + bias;
    __bf16 hh = (__bf16)x;
    Vth[(size_t)n * S + m] = hh;
    Vtl[(size_t)n * S + m] = (__bf16)(x - (float)hh);
  }
}

// ---------------- Kernel 3: QK[h] = Qs_h(64x64) @ wk_h(64x768), store bf16 hi/lo
__global__ void k_qk(const float* __restrict__ Qs, const float* __restrict__ inw,
                     __bf16* __restrict__ QKh, __bf16* __restrict__ QKl) {
  int wid = (blockIdx.x * blockDim.x + threadIdx.x) >> 5;  // 12*4*48 tiles
  int ntile = wid % (H / 16);
  int q = wid / (H / 16);
  int mtile = q & 3;
  int h = q >> 2;
  int m0 = mtile * 16, n0 = ntile * 16;
  const float* wk_h = inw + (size_t)(H + h * DH) * H;
  v8f acc = {0.f, 0.f, 0.f, 0.f, 0.f, 0.f, 0.f, 0.f};
  for (int k0 = 0; k0 < DH; k0 += 32) {
    v16bf ah, al, bh, bl;
    load_a_f32(Qs + h * DH, H, m0, NLP, k0, ah, al);
    load_b_f32_kmaj(wk_h, H, n0, k0, bh, bl);   // B(k=d,n=j) = wk_h[d][j]
    acc = wmma_bf16x3(ah, al, bh, bl, acc);
  }
  int lane = threadIdx.x & 31, g = lane >> 4, nn = lane & 15;
#pragma unroll
  for (int r = 0; r < 8; ++r) {
    int m = m0 + r + (g << 3);
    size_t idx = ((size_t)h * NLP + m) * H + n0 + nn;
    float x = acc[r];
    __bf16 hh = (__bf16)x;
    QKh[idx] = hh;
    QKl[idx] = (__bf16)(x - (float)hh);
  }
}

// ---------------- Kernel 4: logits[h] = QK[h](64x768) @ enc^T
__global__ void k_logits(const __bf16* __restrict__ QKh, const __bf16* __restrict__ QKl,
                         const __bf16* __restrict__ ench, const __bf16* __restrict__ encl,
                         float* __restrict__ lg) {
  int wid = (blockIdx.x * blockDim.x + threadIdx.x) >> 5;  // 12*4*512 tiles
  int stile = wid % (S / 16);
  int q = wid / (S / 16);
  int ltile = q & 3;
  int h = q >> 2;
  int m0 = ltile * 16, n0 = stile * 16;
  const __bf16* Ah = QKh + (size_t)h * NLP * H;
  const __bf16* Al = QKl + (size_t)h * NLP * H;
  v8f acc = {0.f, 0.f, 0.f, 0.f, 0.f, 0.f, 0.f, 0.f};
  for (int k0 = 0; k0 < H; k0 += 32) {
    v16bf ah = load_a_bf(Ah, H, m0, k0);
    v16bf al = load_a_bf(Al, H, m0, k0);
    v16bf bh = load_b_bf(ench, H, n0, k0);   // B(k=j,n=s)=enc[s][j]
    v16bf bl = load_b_bf(encl, H, n0, k0);
    acc = wmma_bf16x3(ah, al, bh, bl, acc);
  }
  int lane = threadIdx.x & 31, g = lane >> 4, nn = lane & 15;
#pragma unroll
  for (int r = 0; r < 8; ++r) {
    int m = m0 + r + (g << 3);
    lg[((size_t)h * NLP + m) * S + n0 + nn] = acc[r];
  }
}

// ---------------- Kernel 5: per-row chunk stats -> prefix (Mpre, Dpre)
__global__ void k_stats(const float* __restrict__ lg, float* __restrict__ Mpre,
                        float* __restrict__ Dpre) {
  int row = blockIdx.x;                       // h*64 + l, 768 rows
  const float* lp = lg + (size_t)row * S;
  __shared__ float Mc[NC], Sc[NC];
  int t = threadIdx.x;
  if (t < NC) {
    const float* p = lp + t * T;
    float m = -3.0e38f;
    for (int i = 0; i < T; ++i) m = fmaxf(m, p[i]);
    float s = 0.f;
    for (int i = 0; i < T; ++i) s += __expf(p[i] - m);
    Mc[t] = m; Sc[t] = s;
  }
  __syncthreads();
  if (t == 0) {
    float M = -3.0e38f, D = 0.f;
    for (int b = 0; b < NC; ++b) {
      float m2 = fmaxf(M, Mc[b]);
      D = D * __expf(M - m2) + Sc[b] * __expf(Mc[b] - m2);
      M = m2;
      Mpre[(size_t)row * NC + b] = M;
      Dpre[(size_t)row * NC + b] = D;
    }
  }
}

// ---------------- Kernel 6: ctx[b,l,h*64+d] = softmax_prefix(logits) @ V
__global__ void k_pv(const float* __restrict__ lg,
                     const __bf16* __restrict__ Vth, const __bf16* __restrict__ Vtl,
                     const float* __restrict__ Mpre, const float* __restrict__ Dpre,
                     float* __restrict__ ctx) {
  int wid = (blockIdx.x * blockDim.x + threadIdx.x) >> 5;  // 16*12*4*4 waves
  int dtile = wid & 3;
  int q = wid >> 2;
  int ltile = q & 3;  q >>= 2;
  int h = q % NH;
  int b = q / NH;
  int lane = threadIdx.x & 31, g = lane >> 4;

  int l_a = ltile * 16 + (lane & 15);
  int row = h * NLP + l_a;
  float mrow = Mpre[(size_t)row * NC + b];
  const float* lp = lg + (size_t)row * S;
  const __bf16* Bh = Vth + (size_t)h * DH * S;
  const __bf16* Bl = Vtl + (size_t)h * DH * S;

  v8f acc = {0.f, 0.f, 0.f, 0.f, 0.f, 0.f, 0.f, 0.f};
  int smax = (b + 1) * T;
  for (int s0 = 0; s0 < smax; s0 += 32) {
    // A = exp(logit - Mpre), split on the fly (elem run*8+j <-> K = g*16 + run*8 + j)
    v16bf ah, al;
    const float* p0 = lp + s0 + (g << 4);
#pragma unroll
    for (int run = 0; run < 2; ++run) {
      const float* p = p0 + run * 8;
      float4 a = *(const float4*)p;
      float4 c2 = *(const float4*)(p + 4);
      float v[8] = {a.x, a.y, a.z, a.w, c2.x, c2.y, c2.z, c2.w};
#pragma unroll
      for (int j = 0; j < 8; ++j) {
        float e = __expf(v[j] - mrow);
        __bf16 hh = (__bf16)e;
        ah[run * 8 + j] = hh;
        al[run * 8 + j] = (__bf16)(e - (float)hh);
      }
    }
    v16bf bh = load_b_bf(Bh, S, dtile * 16, s0);  // B(k=s,n=d)=Vt[d][s]
    v16bf bl = load_b_bf(Bl, S, dtile * 16, s0);
    acc = wmma_bf16x3(ah, al, bh, bl, acc);
  }
  int d = h * DH + dtile * 16 + (lane & 15);
#pragma unroll
  for (int r = 0; r < 8; ++r) {
    int l = ltile * 16 + r + (g << 3);
    float D = Dpre[((size_t)h * NLP + l) * NC + b];
    ctx[((size_t)b * NLP + l) * H + d] = acc[r] / D;
  }
}

// ---------------- Kernel 7: wt = lw(50x768) @ out_w(768x768)
__global__ void k_wt(const float* __restrict__ lw, const float* __restrict__ ow,
                     float* __restrict__ wt) {
  int wid = (blockIdx.x * blockDim.x + threadIdx.x) >> 5;  // 4*48 tiles
  int ntile = wid % (H / 16);
  int mtile = wid / (H / 16);
  int m0 = mtile * 16, n0 = ntile * 16;
  v8f acc = {0.f, 0.f, 0.f, 0.f, 0.f, 0.f, 0.f, 0.f};
  for (int k0 = 0; k0 < H; k0 += 32) {
    v16bf ah, al, bh, bl;
    load_a_f32(lw, H, m0, NL, k0, ah, al);
    load_b_f32_kmaj(ow, H, n0, k0, bh, bl);   // B(k=jo,n=i)=out_w[jo][i]
    acc = wmma_bf16x3(ah, al, bh, bl, acc);
  }
  int lane = threadIdx.x & 31, g = lane >> 4, nn = lane & 15;
#pragma unroll
  for (int r = 0; r < 8; ++r) {
    int m = m0 + r + (g << 3);
    wt[(size_t)m * H + n0 + nn] = acc[r];
  }
}

// ---------------- Kernel 8: score[b,l] = ctx[b,l,:]·wt[l,:] + lw[l,:]·out_b
__global__ void k_score(const float* __restrict__ ctx, const float* __restrict__ wt,
                        const float* __restrict__ lw, const float* __restrict__ outb,
                        float* __restrict__ out) {
  int tid = blockIdx.x * blockDim.x + threadIdx.x;
  if (tid >= NC * NL) return;
  int b = tid / NL, l = tid % NL;
  const float* c = ctx + ((size_t)b * NLP + l) * H;
  const float* w = wt + (size_t)l * H;
  const float* q = lw + (size_t)l * H;
  float s = 0.f;
  for (int i = 0; i < H; ++i) s = fmaf(c[i], w[i], s);
  float sb = 0.f;
  for (int i = 0; i < H; ++i) sb = fmaf(q[i], outb[i], sb);
  out[(size_t)b * NL + l] = s + sb;
}

extern "C" void kernel_launch(void* const* d_in, const int* in_sizes, int n_in,
                              void* d_out, int out_size, void* d_ws, size_t ws_size,
                              hipStream_t stream) {
  const float* enc  = (const float*)d_in[0];
  const float* lq   = (const float*)d_in[1];
  const float* lw   = (const float*)d_in[2];   // (H,NL) buffer, used flat as (NL,H)
  const float* inw  = (const float*)d_in[3];
  const float* inb  = (const float*)d_in[4];
  const float* outw = (const float*)d_in[5];
  const float* outb = (const float*)d_in[6];
  float* out = (float*)d_out;
  char* ws = (char*)d_ws;

  (void)in_sizes; (void)n_in; (void)out_size; (void)ws_size;

  __bf16* ench = (__bf16*)(ws + B_ENCH);
  __bf16* encl = (__bf16*)(ws + B_ENCL);
  __bf16* wvh  = (__bf16*)(ws + B_WVH);
  __bf16* wvl  = (__bf16*)(ws + B_WVL);
  float*  Qs   = (float*)(ws + B_QS);
  __bf16* QKh  = (__bf16*)(ws + B_QKH);
  __bf16* QKl  = (__bf16*)(ws + B_QKL);
  __bf16* Vth  = (__bf16*)(ws + B_VTH);
  __bf16* Vtl  = (__bf16*)(ws + B_VTL);
  float*  lg   = (float*)(ws + B_LG);
  float*  Mp   = (float*)(ws + B_MP);
  float*  Dp   = (float*)(ws + B_DP);
  float*  ctx  = (float*)(ws + B_CTX);
  float*  wt   = (float*)(ws + B_WT);

  // split f32 operands into bf16 hi/lo once
  k_split4<<<(S * H / 4 + 255) / 256, 256, 0, stream>>>(enc, ench, encl, S * H / 4);
  k_split4<<<(H * H / 4 + 255) / 256, 256, 0, stream>>>(inw + (size_t)2 * H * H,
                                                        wvh, wvl, H * H / 4);

  k_qproj <<<  24, 256, 0, stream>>>(lq, inw, inb, Qs);
  k_vproj <<<3072, 256, 0, stream>>>(ench, encl, wvh, wvl, inb, Vth, Vtl);
  k_qk    <<< 288, 256, 0, stream>>>(Qs, inw, QKh, QKl);
  k_logits<<<3072, 256, 0, stream>>>(QKh, QKl, ench, encl, lg);
  k_stats <<< 768,  64, 0, stream>>>(lg, Mp, Dp);
  k_pv    <<< 384, 256, 0, stream>>>(lg, Vth, Vtl, Mp, Dp, ctx);
  k_wt    <<<  24, 256, 0, stream>>>(lw, outw, wt);
  k_score <<<   4, 256, 0, stream>>>(ctx, wt, lw, outb, out);
}
